// QFCHybridModel_65481071399775
// MI455X (gfx1250) — compile-verified, matrix operand order checked by reading, not run on hardware
//
#include <hip/hip_runtime.h>
#include <hip/hip_bf16.h>
#include <math.h>

typedef __bf16 bf16_t;
typedef __bf16 v16bf __attribute__((ext_vector_type(16)));
typedef __bf16 v8bf  __attribute__((ext_vector_type(8)));
typedef float  v8f   __attribute__((ext_vector_type(8)));
typedef unsigned int u32x4 __attribute__((ext_vector_type(4)));
typedef int          i32x8 __attribute__((ext_vector_type(8)));
typedef int          i32x4 __attribute__((ext_vector_type(4)));

#define C1 8
#define C2 16
#define H1 64

#if defined(__has_builtin)
#  if __has_builtin(__builtin_amdgcn_tensor_load_to_lds)
#    define USE_TDM 1
#  endif
#endif
#ifndef USE_TDM
#  define USE_TDM 0
#endif

// ---------------------------------------------------------------------------
// Kernel 1: conv1 (1->8ch, 3x3 SAME) + relu + 2x2 maxpool, emit bf16 NHWC.
// One block per (image, 8-row pooled strip). LDS-tiled input with halo.
// ---------------------------------------------------------------------------
__global__ __launch_bounds__(256)
void conv1_relu_pool_kernel(const float* __restrict__ x,
                            const float* __restrict__ w1,
                            const float* __restrict__ b1,
                            bf16_t* __restrict__ h1p)   // [512][64][64][8] NHWC
{
    __shared__ float sX[18][130];
    __shared__ float sW[8][9];
    __shared__ float sB[8];

    const int blk   = blockIdx.x;
    const int b     = blk >> 3;
    const int strip = blk & 7;
    const int tid   = threadIdx.x;

    if (tid < 72) sW[tid / 9][tid % 9] = w1[tid];
    if (tid < 8)  sB[tid] = b1[tid];

    const int row0 = strip * 16 - 1;
    const float* xb = x + (size_t)b * 128 * 128;
    for (int idx = tid; idx < 18 * 130; idx += 256) {
        int r = idx / 130, c = idx % 130;
        int gr = row0 + r, gc = c - 1;
        float v = 0.f;
        if ((unsigned)gr < 128u && (unsigned)gc < 128u)
            v = xb[gr * 128 + gc];
        sX[r][c] = v;
    }
    // gfx1250 prefetch path: pull the next strip toward the caches
    if (strip < 7) {
        const float* nxt = xb + (size_t)(strip * 16 + 16) * 128 + (tid & 31) * 16;
        __builtin_prefetch(nxt, 0, 1);
    }
    __syncthreads();

    // 8 pooled rows x 64 pooled cols x 8 channels; ch innermost => NHWC stores coalesce
    for (int t = tid; t < 4096; t += 256) {
        int ch  = t & 7;
        int pos = t >> 3;
        int pr  = pos >> 6, pc = pos & 63;
        float m = 0.f;  // relu makes all candidates >= 0
        #pragma unroll
        for (int sy = 0; sy < 2; ++sy) {
            #pragma unroll
            for (int sx2 = 0; sx2 < 2; ++sx2) {
                int rr = pr * 2 + sy;
                int cc = pc * 2 + sx2;
                float acc = sB[ch];
                #pragma unroll
                for (int dy = 0; dy < 3; ++dy)
                    #pragma unroll
                    for (int dx = 0; dx < 3; ++dx)
                        acc += sX[rr + dy][cc + dx] * sW[ch][dy * 3 + dx];
                acc = fmaxf(acc, 0.f);
                m = fmaxf(m, acc);
            }
        }
        size_t off = (((size_t)b * H1 + (strip * 8 + pr)) * H1 + pc) * C1 + ch;
        h1p[off] = (bf16_t)m;
    }
}

// ---------------------------------------------------------------------------
// Helpers for kernel 2
// ---------------------------------------------------------------------------
__device__ __forceinline__ unsigned lds_byte_off(const void* p) {
    // Flat LDS addresses carry the workgroup LDS byte offset in the low 32 bits
    // (aperture bits live in [63:32] per the CDNA5 aperture scheme).
    return (unsigned)(unsigned long long)p;
}

// VALU fallback / top-edge staging: rows gstart..gstart+3 into buf, cols 1..64,
// zero for out-of-range rows. Never touches halo cols 0/65 (stay zero).
__device__ __forceinline__ void stage_rows_valu(bf16_t (*buf)[66][8],
                                                const bf16_t* __restrict__ hb,
                                                int gstart, int tid) {
    for (int idx = tid; idx < 4 * 64 * 8; idx += 256) {
        int lr  = idx >> 9;           // /512
        int rem = idx & 511;
        int col = rem >> 3, ic = rem & 7;
        int gr = gstart + lr;
        bf16_t v = (bf16_t)0.f;
        if ((unsigned)gr < 64u)
            v = hb[((size_t)gr * 64 + col) * 8 + ic];
        buf[lr][col + 1][ic] = v;
    }
}

#if USE_TDM
// TDM 2D descriptor: tile = 4 rows x 512 elems (64 cols x 8 ch, bf16),
// LDS pad of 8 DWORDs after every 256 DWORDs skips the halo channel-groups,
// giving an LDS row pitch of 66*8 bf16. tensor_dim1 = rows remaining from the
// tile start enables bottom-edge zero fill.
__device__ __forceinline__ void tdm_load_rows(const bf16_t* __restrict__ gsrc,
                                              unsigned lds_off, int rows_remaining) {
    unsigned long long ga = (unsigned long long)gsrc;
    u32x4 g0;
    g0[0] = 1u;                                            // count = 1 (valid D#)
    g0[1] = lds_off;                                       // LDS dest (bytes)
    g0[2] = (unsigned)(ga & 0xffffffffull);                // global_addr lo
    g0[3] = (unsigned)((ga >> 32) & 0x01ffffffull) | (2u << 30);  // hi | type=2
    i32x8 g1;
    g1[0] = (1 << 16) | (1 << 20) | (7 << 22) | (7 << 25); // data_size=2B, pad_en,
                                                           // interval=256DW, amount=8DW
    g1[1] = (512 << 16);                                   // tensor_dim0 = 512
    g1[2] = (rows_remaining & 0xffff) << 16;               // tensor_dim1
    g1[3] = (512 << 16);                                   // tile_dim0 = 512
    g1[4] = 4;                                             // tile_dim1 = 4, dim2 unused
    g1[5] = 512;                                           // tensor_dim0_stride
    g1[6] = (512 << 16);                                   // tensor_dim1_stride lo16
    g1[7] = 0;
    i32x4 g2 = {0, 0, 0, 0};
    i32x4 g3 = {0, 0, 0, 0};
#if __clang_major__ >= 23
    i32x8 g4 = {0, 0, 0, 0, 0, 0, 0, 0};
    __builtin_amdgcn_tensor_load_to_lds(g0, g1, g2, g3, g4, 0);
#else
    __builtin_amdgcn_tensor_load_to_lds(g0, g1, g2, g3, 0);
#endif
}
#endif

// One 16-byte A-fragment quarter: 8 consecutive channels at tap t (dy,dx).
// Invalid taps (t > 8, the zero-padded K region) read a zeroed 16-B LDS slot.
__device__ __forceinline__ v8bf tap_load(const bf16_t (*buf)[66][8], int rsel,
                                         int colA, int t, const bf16_t* zp) {
    int tc = (t <= 8) ? t : 0;
    int dy = tc / 3, dx = tc - dy * 3;
    const bf16_t* p = (t <= 8) ? &buf[rsel + dy][colA + dx][0] : zp;
    return *(const v8bf*)p;
}

// ---------------------------------------------------------------------------
// Kernel 2: conv2 (8->16ch, 3x3 SAME) as implicit GEMM on bf16 WMMA
//   M = 16 positions, N = 16 out channels, K = 72 (tap*8+ic, 3 chunks of 32)
// TDM double-buffered LDS staging, fused relu + 2x2 pool + global mean.
// One block (8 waves) per image; each iteration = 2 output rows.
// ---------------------------------------------------------------------------
__global__ __launch_bounds__(256)
void conv2_wmma_pool_mean_kernel(const bf16_t* __restrict__ h1p,  // NHWC
                                 const float* __restrict__ w2,
                                 const float* __restrict__ b2,
                                 float* __restrict__ feats)
{
    __shared__ __align__(16) bf16_t sIn[2][4][66][8]; // rows r-1..r+2, halo, ch-inner
    __shared__ bf16_t sBmat[96][16];                  // K x N weights (K zero-padded)
    __shared__ float  sOut[2][64][16];                // conv2 row pair [row][col][ch]
    __shared__ float  sBias[16];
    __shared__ float  sRed[16][16];
    __shared__ __align__(16) bf16_t sZero[8];

    const int tid  = threadIdx.x;
    const int lane = tid & 31;
    const int wv   = tid >> 5;
    const int b    = blockIdx.x;

    // Stage B with tap-major K: k = tap*8 + ic, element = w2[n][ic][tap]
    for (int idx = tid; idx < 96 * 16; idx += 256) {
        int k = idx >> 4, n = idx & 15;
        float v = 0.f;
        if (k < 72) v = w2[(n * 8 + (k & 7)) * 9 + (k >> 3)];
        sBmat[k][n] = (bf16_t)v;
    }
    if (tid < 16) sBias[tid] = b2[tid];
    for (int idx = tid; idx < 2 * 4 * 66 * 8; idx += 256)
        ((bf16_t*)sIn)[idx] = (bf16_t)0.f;            // halos stay zero forever
    if (tid < 8) sZero[tid] = (bf16_t)0.f;
    __syncthreads();

    // Loop-invariant B fragments (16-bit B layout mirrors A: lanes 0-15 hold
    // K{kb..kb+7, 16+kb..}, kb=0; lanes 16-31 kb=8; N = lane%16)
    const int kbase = (lane < 16) ? 0 : 8;
    const int nB    = lane & 15;
    v16bf bfrag[3];
    #pragma unroll
    for (int kc = 0; kc < 3; ++kc) {
        #pragma unroll
        for (int e = 0; e < 16; ++e) {
            int K = kbase + e + ((e >= 8) ? 8 : 0);
            bfrag[kc][e] = sBmat[kc * 32 + K][nB];
        }
    }

    const bf16_t* hb = h1p + (size_t)b * H1 * H1 * C1;
    const int rsel    = wv >> 2;
    const int colbase = (wv & 3) * 16;
    const int mA      = lane & 15;
    const int colA    = colbase + mA;
    const int hi1     = (lane >> 4) & 1;

    const int ch  = tid & 15;
    const int grp = tid >> 4;
    float acc = 0.f;

#if USE_TDM
    stage_rows_valu(sIn[0], hb, -1, tid);   // top edge: row -1 stays zero
    __syncthreads();
#endif

    for (int i = 0; i < 32; ++i) {
        const int r = 2 * i;
#if USE_TDM
        const int cur = i & 1;
        if (r < 62 && wv == 0)              // async-prefetch next row pair
            tdm_load_rows(hb + (size_t)(r + 1) * 512,
                          lds_byte_off(&sIn[cur ^ 1][0][1][0]), 63 - r);
#else
        const int cur = 0;
        stage_rows_valu(sIn[0], hb, r - 1, tid);
        __syncthreads();
#endif
        const bf16_t (*buf)[66][8] = sIn[cur];

        // Implicit-GEMM tile: 3 K-chunks, A fragment = two ds_load_b128 per chunk
        v8f c8 = {};
        #pragma unroll
        for (int kc = 0; kc < 3; ++kc) {
            int t0 = 4 * kc + hi1;
            v8bf lo = tap_load(buf, rsel, colA, t0,     sZero);
            v8bf hi = tap_load(buf, rsel, colA, t0 + 2, sZero);
            v16bf afrag = __builtin_shufflevector(lo, hi,
                0, 1, 2, 3, 4, 5, 6, 7, 8, 9, 10, 11, 12, 13, 14, 15);
            c8 = __builtin_amdgcn_wmma_f32_16x16x32_bf16(
                     false, afrag, false, bfrag[kc], (short)0, c8, false, false);
        }

        // bias + relu, scatter C tile (lane<16: N=lane, M=v; lane>=16: M=v+8)
        {
            int nC    = lane & 15;
            int mbase = (lane < 16) ? 0 : 8;
            #pragma unroll
            for (int v = 0; v < 8; ++v) {
                float y = fmaxf(c8[v] + sBias[nC], 0.f);
                sOut[rsel][colbase + mbase + v][nC] = y;
            }
        }
        __syncthreads();

        // 2x2 max pool over the row pair + mean accumulation
        #pragma unroll
        for (int g = 0; g < 2; ++g) {
            int pc = grp + g * 16;
            float p0 = sOut[0][2 * pc    ][ch];
            float p1 = sOut[0][2 * pc + 1][ch];
            float p2 = sOut[1][2 * pc    ][ch];
            float p3 = sOut[1][2 * pc + 1][ch];
            acc += fmaxf(fmaxf(p0, p1), fmaxf(p2, p3));
        }
#if USE_TDM
        if (r < 62 && wv == 0) __builtin_amdgcn_s_wait_tensorcnt(0);
#endif
        __syncthreads();
    }

    sRed[grp][ch] = acc;
    __syncthreads();
    if (tid < 4) {  // only channels 0..3 feed the quantum circuit
        float s = 0.f;
        #pragma unroll
        for (int g = 0; g < 16; ++g) s += sRed[g][tid];
        feats[b * 4 + tid] = s * (1.0f / 1024.0f);
    }
}

// ---------------------------------------------------------------------------
// Kernel 3: 4-wire quantum circuit + batchnorm over batch + 4x4 head.
// ---------------------------------------------------------------------------
__global__ __launch_bounds__(512)
void quantum_bn_head_kernel(const float* __restrict__ feats,
                            const float* __restrict__ q_params,
                            const float* __restrict__ bn_gamma,
                            const float* __restrict__ bn_beta,
                            const float* __restrict__ head_w,
                            const float* __restrict__ head_b,
                            float* __restrict__ out)
{
    __shared__ float sSum[4], sSq[4], sMu[4], sInvStd[4];
    const int tid = threadIdx.x;
    if (tid < 4) { sSum[tid] = 0.f; sSq[tid] = 0.f; }
    __syncthreads();

    float a0r[4], a0i[4], a1r[4], a1i[4];
    #pragma unroll
    for (int i = 0; i < 4; ++i) {
        float hf = 0.5f * feats[tid * 4 + i];
        float hp = 0.5f * q_params[i];
        float cy = __cosf(hf), sy = __sinf(hf);
        float cx = __cosf(hp), sx = __sinf(hp);
        a0r[i] = cx * cy;  a0i[i] = -sx * sy;
        a1r[i] = cx * sy;  a1i[i] = -sx * cy;
    }
    float pr[16], pi[16];
    #pragma unroll
    for (int s = 0; s < 16; ++s) {
        float re = 1.f, im = 0.f;
        #pragma unroll
        for (int i = 0; i < 4; ++i) {
            int bit = (s >> (3 - i)) & 1;
            float br = bit ? a1r[i] : a0r[i];
            float bi = bit ? a1i[i] : a0i[i];
            float nr = re * br - im * bi;
            float ni = re * bi + im * br;
            re = nr; im = ni;
        }
        pr[s] = re; pi[s] = im;
    }
    #pragma unroll
    for (int c = 0; c < 3; ++c) {
        float tr[16], ti[16];
        #pragma unroll
        for (int s = 0; s < 16; ++s) {
            int cbit = (s >> (3 - c)) & 1;
            int src  = cbit ? (s ^ (1 << (2 - c))) : s;
            tr[s] = pr[src]; ti[s] = pi[src];
        }
        #pragma unroll
        for (int s = 0; s < 16; ++s) { pr[s] = tr[s]; pi[s] = ti[s]; }
    }
    float q[4] = {0.f, 0.f, 0.f, 0.f};
    #pragma unroll
    for (int s = 0; s < 16; ++s) {
        float p = pr[s] * pr[s] + pi[s] * pi[s];
        #pragma unroll
        for (int i = 0; i < 4; ++i)
            q[i] += p * (((s >> (3 - i)) & 1) ? -1.f : 1.f);
    }
    #pragma unroll
    for (int i = 0; i < 4; ++i) {
        atomicAdd(&sSum[i], q[i]);
        atomicAdd(&sSq[i],  q[i] * q[i]);
    }
    __syncthreads();
    if (tid < 4) {
        float mu  = sSum[tid] * (1.f / 512.f);
        float var = sSq[tid] * (1.f / 512.f) - mu * mu;
        sMu[tid]     = mu;
        sInvStd[tid] = rsqrtf(var + 1e-5f);
    }
    __syncthreads();
    float nrm[4];
    #pragma unroll
    for (int i = 0; i < 4; ++i)
        nrm[i] = bn_gamma[i] * (q[i] - sMu[i]) * sInvStd[i] + bn_beta[i];
    #pragma unroll
    for (int j = 0; j < 4; ++j) {
        float o = head_b[j];
        #pragma unroll
        for (int i = 0; i < 4; ++i)
            o += nrm[i] * head_w[j * 4 + i];
        out[tid * 4 + j] = o;
    }
}

// ---------------------------------------------------------------------------
extern "C" void kernel_launch(void* const* d_in, const int* in_sizes, int n_in,
                              void* d_out, int out_size, void* d_ws, size_t ws_size,
                              hipStream_t stream) {
    (void)in_sizes; (void)n_in; (void)out_size; (void)ws_size;
    const float* x   = (const float*)d_in[0];
    const float* w1  = (const float*)d_in[1];
    const float* b1  = (const float*)d_in[2];
    const float* w2  = (const float*)d_in[3];
    const float* b2  = (const float*)d_in[4];
    const float* qp  = (const float*)d_in[5];
    const float* gmm = (const float*)d_in[6];
    const float* bta = (const float*)d_in[7];
    const float* hw  = (const float*)d_in[8];
    const float* hb  = (const float*)d_in[9];
    float* out = (float*)d_out;

    // ws: [0, 8KB) feats (512x4 f32); [8KB, +32MB) h1p bf16 NHWC (512x64x64x8)
    float*  feats = (float*)d_ws;
    bf16_t* h1p   = (bf16_t*)((char*)d_ws + 8192);

    conv1_relu_pool_kernel<<<512 * 8, 256, 0, stream>>>(x, w1, b1, h1p);
    conv2_wmma_pool_mean_kernel<<<512, 256, 0, stream>>>(h1p, w2, b2, feats);
    quantum_bn_head_kernel<<<1, 512, 0, stream>>>(feats, qp, gmm, bta, hw, hb, out);
}
